// TopKSAE_16106127360071
// MI455X (gfx1250) — matrix operand order repaired and checked by reading, not compiled, not run on hardware
//
#include <hip/hip_runtime.h>
#include <hip/hip_bf16.h>
#include <stdint.h>
#include <stddef.h>

#define INPUT_DIM   2048
#define LATENT_DIM  16384
#define BATCH       8192
#define TOPK        32
#define MAX_NNZ     128

#define BM 128
#define BN 128
#define BK 32
#define BKP 40   // padded row stride (bf16 elems) to avoid LDS bank conflicts
#define KSTEPS (INPUT_DIM / BK)   // 64

typedef __attribute__((ext_vector_type(16))) __bf16 v16bf;
typedef __attribute__((ext_vector_type(8)))  float  v8f;
typedef __attribute__((ext_vector_type(4)))  float  v4f;
typedef __attribute__((ext_vector_type(4)))  int    v4i;

union FragBF { uint4 u[2]; v16bf v; };

// --------- CDNA5 async global->LDS path (guarded; fallback = sync staging) ---
#if defined(__has_builtin)
#if __has_builtin(__builtin_amdgcn_global_load_async_to_lds_b128) && \
    __has_builtin(__builtin_amdgcn_s_wait_asynccnt)
#define HAVE_ASYNC_LDS 1
#endif
#endif

#ifdef HAVE_ASYNC_LDS
typedef __attribute__((address_space(1))) v4i GVi;   // global
typedef __attribute__((address_space(3))) v4i LVi;   // LDS
__device__ __forceinline__ void async_copy_b128(const void* g, void* l) {
    // global flat addr == AS1 addr; LDS aperture: low 32 bits == LDS offset.
    __builtin_amdgcn_global_load_async_to_lds_b128(
        (GVi*)(unsigned long long)(uintptr_t)g,
        (LVi*)(unsigned)(uintptr_t)l, 0, 0);
}
#endif

// ---------------------------------------------------------------------------
// 1) x (f32) -> x_bf16
// ---------------------------------------------------------------------------
__global__ __launch_bounds__(256) void k_convert_x(const float* __restrict__ x,
                                                   __bf16* __restrict__ xb) {
    size_t i = ((size_t)blockIdx.x * 256 + threadIdx.x) * 4;
    v4f f = __builtin_nontemporal_load((const v4f*)(x + i));  // read-once
    union { __bf16 b[4]; uint2 u; } p;
    p.b[0] = (__bf16)f.x; p.b[1] = (__bf16)f.y;
    p.b[2] = (__bf16)f.z; p.b[3] = (__bf16)f.w;
    *(uint2*)(xb + i) = p.u;   // RT: reused by GEMM, keep cached
}

// ---------------------------------------------------------------------------
// 2) W_enc [K=2048][N=16384] f32 -> W_encT [N][K] bf16 (LDS-staged transpose)
// ---------------------------------------------------------------------------
__global__ __launch_bounds__(256) void k_convT_w(const float* __restrict__ W,
                                                 __bf16* __restrict__ Wt) {
    __shared__ __bf16 tile[64][72];
    int n0 = blockIdx.x * 64;
    int k0 = blockIdx.y * 64;
    int t  = threadIdx.x;
#pragma unroll
    for (int i = 0; i < 16; ++i) {
        int e = t + i * 256;
        int r = e >> 6;      // k within tile
        int c = e & 63;      // n within tile
        tile[c][r] = (__bf16)__builtin_nontemporal_load(
                         &W[(size_t)(k0 + r) * LATENT_DIM + n0 + c]);
    }
    __syncthreads();
#pragma unroll
    for (int i = 0; i < 16; ++i) {
        int e = t + i * 256;
        int r = e >> 6;      // n within tile
        int c = e & 63;      // k within tile
        Wt[(size_t)(n0 + r) * INPUT_DIM + k0 + c] = tile[r][c];
    }
}

// ---------------------------------------------------------------------------
// 3) Encoder GEMM: latent = relu(x @ W_enc + b_enc), bf16 WMMA, f32 accum.
//    Double-buffered LDS; async DMA global->LDS when the toolchain exposes it.
// ---------------------------------------------------------------------------
__global__ __launch_bounds__(256) void k_encoder_gemm(
        const __bf16* __restrict__ xb,   // [BATCH][INPUT_DIM]
        const __bf16* __restrict__ wt,   // [LATENT_DIM][INPUT_DIM]
        const float*  __restrict__ b_enc,
        float*        __restrict__ latent) {
    __shared__ __attribute__((aligned(16))) __bf16 sA[2][BM * BKP];
    __shared__ __attribute__((aligned(16))) __bf16 sB[2][BN * BKP];

    const int t  = threadIdx.x;
    const int m0 = blockIdx.y * BM;
    const int n0 = blockIdx.x * BN;

    const int wid = t >> 5;          // wave 0..7
    const int l   = t & 31;
    const int l16 = l & 15;
    const int lh  = l >> 4;          // half-wave 0/1
    const int wm  = (wid & 3) * 32;  // wave row base within block
    const int wn  = (wid >> 2) * 64; // wave col base within block

    v8f acc[2][4];
#pragma unroll
    for (int i = 0; i < 2; ++i)
#pragma unroll
        for (int j = 0; j < 4; ++j)
#pragma unroll
            for (int r = 0; r < 8; ++r) acc[i][j][r] = 0.f;

    // staging: each thread moves 32B of A and 32B of B per K-step
    const int lrow = t >> 1;           // 0..127
    const int lkc  = (t & 1) * 16;     // 0 or 16
    const __bf16* gA = xb + (size_t)(m0 + lrow) * INPUT_DIM + lkc;
    const __bf16* gB = wt + (size_t)(n0 + lrow) * INPUT_DIM + lkc;
    __bf16* lA0 = &sA[0][lrow * BKP + lkc];
    __bf16* lA1 = &sA[1][lrow * BKP + lkc];
    __bf16* lB0 = &sB[0][lrow * BKP + lkc];
    __bf16* lB1 = &sB[1][lrow * BKP + lkc];

#ifdef HAVE_ASYNC_LDS
    // ---- CDNA5 async pipeline: DMA tile k+1 while WMMAs chew tile k --------
    {
        async_copy_b128(gA,     lA0);
        async_copy_b128(gA + 8, lA0 + 8);
        async_copy_b128(gB,     lB0);
        async_copy_b128(gB + 8, lB0 + 8);
    }
    for (int it = 0; it < KSTEPS; ++it) {
        const int cur = it & 1;
        if (it + 1 < KSTEPS) {
            const __bf16* pa = gA + (size_t)(it + 1) * BK;
            const __bf16* pb = gB + (size_t)(it + 1) * BK;
            __bf16* la = cur ? lA0 : lA1;
            __bf16* lb = cur ? lB0 : lB1;
            async_copy_b128(pa,     la);
            async_copy_b128(pa + 8, la + 8);
            async_copy_b128(pb,     lb);
            async_copy_b128(pb + 8, lb + 8);
            __builtin_amdgcn_s_wait_asynccnt(4);  // oldest 4 (== cur tile) done
        } else {
            __builtin_amdgcn_s_wait_asynccnt(0);
        }
        __syncthreads();

        const __bf16* bufA = sA[cur];
        const __bf16* bufB = sB[cur];
        FragBF fa[2];
#pragma unroll
        for (int i = 0; i < 2; ++i) {
            const uint4* p = (const uint4*)(bufA + (wm + i * 16 + l16) * BKP + lh * 16);
            fa[i].u[0] = p[0];
            fa[i].u[1] = p[1];
        }
        FragBF fb[4];
#pragma unroll
        for (int j = 0; j < 4; ++j) {
            const uint4* p = (const uint4*)(bufB + (wn + j * 16 + l16) * BKP + lh * 16);
            fb[j].u[0] = p[0];
            fb[j].u[1] = p[1];
        }
#pragma unroll
        for (int i = 0; i < 2; ++i)
#pragma unroll
            for (int j = 0; j < 4; ++j)
                acc[i][j] = __builtin_amdgcn_wmma_f32_16x16x32_bf16(
                    false, fa[i].v, false, fb[j].v,
                    (short)0, acc[i][j], false, false);
        __syncthreads();
    }
#else
    // ---- fallback: sync staging through VGPRs ----------------
    const uint4* pa = (const uint4*)gA;
    const uint4* pb = (const uint4*)gB;
    uint4* sAp = (uint4*)lA0;
    uint4* sBp = (uint4*)lB0;
    for (int k0 = 0; k0 < INPUT_DIM; k0 += BK) {
        uint4 a0 = pa[0], a1 = pa[1];
        uint4 b0 = pb[0], b1 = pb[1];
        __syncthreads();
        sAp[0] = a0; sAp[1] = a1;
        sBp[0] = b0; sBp[1] = b1;
        __syncthreads();

        FragBF fa[2];
#pragma unroll
        for (int i = 0; i < 2; ++i) {
            const uint4* p = (const uint4*)(&sA[0][0] + (wm + i * 16 + l16) * BKP + lh * 16);
            fa[i].u[0] = p[0];
            fa[i].u[1] = p[1];
        }
        FragBF fb[4];
#pragma unroll
        for (int j = 0; j < 4; ++j) {
            const uint4* p = (const uint4*)(&sB[0][0] + (wn + j * 16 + l16) * BKP + lh * 16);
            fb[j].u[0] = p[0];
            fb[j].u[1] = p[1];
        }
#pragma unroll
        for (int i = 0; i < 2; ++i)
#pragma unroll
            for (int j = 0; j < 4; ++j)
                acc[i][j] = __builtin_amdgcn_wmma_f32_16x16x32_bf16(
                    false, fa[i].v, false, fb[j].v,
                    (short)0, acc[i][j], false, false);
        pa += 4;
        pb += 4;
    }
#endif

    // bias + relu + store (C layout: VGPR r -> rows r and r+8, col = lane%16)
    float bias[4];
#pragma unroll
    for (int j = 0; j < 4; ++j) bias[j] = b_enc[n0 + wn + j * 16 + l16];
#pragma unroll
    for (int i = 0; i < 2; ++i) {
        const int rowb = m0 + wm + i * 16 + lh * 8;
#pragma unroll
        for (int j = 0; j < 4; ++j) {
            const int col = n0 + wn + j * 16 + l16;
            float* out = latent + (size_t)rowb * LATENT_DIM + col;
#pragma unroll
            for (int r = 0; r < 8; ++r) {
                float v = acc[i][j][r] + bias[j];
                // latent is 512MB write-once/read-once: keep it out of L2
                __builtin_nontemporal_store(v > 0.f ? v : 0.f,
                                            out + (size_t)r * LATENT_DIM);
            }
        }
    }
}

// ---------------------------------------------------------------------------
// 4) Per-row exact top-k threshold (binary search on float bit patterns;
//    ReLU output >= 0 so uint order == float order), in-place masking, and
//    deterministic compaction of survivors for the sparse decoder.
// ---------------------------------------------------------------------------
__global__ __launch_bounds__(256) void k_topk(float* __restrict__ latent,
                                              int*   __restrict__ counts,
                                              int*   __restrict__ idxs,
                                              float* __restrict__ vals) {
    const int row = blockIdx.x;
    const int t   = threadIdx.x;
    float* lrow = latent + (size_t)row * LATENT_DIM;

    unsigned u[64];
#pragma unroll
    for (int i = 0; i < 64; ++i)
        u[i] = __float_as_uint(__builtin_nontemporal_load(lrow + t + i * 256));

    __shared__ int s_cnt;
    unsigned lo = 0u, hi = 0x7f7fffffu;   // [0, max finite]
    while (lo < hi) {                      // ~31 uniform iterations
        unsigned mid = lo + ((hi - lo + 1u) >> 1);
        if (t == 0) s_cnt = 0;
        __syncthreads();
        int c = 0;
#pragma unroll
        for (int i = 0; i < 64; ++i) c += (u[i] >= mid);
        atomicAdd(&s_cnt, c);
        __syncthreads();
        int total = s_cnt;
        __syncthreads();
        if (total >= TOPK) lo = mid; else hi = mid - 1u;
    }
    const unsigned thr = lo;   // == 32nd-largest value's bits (0 if <32 positives)

    // deterministic compaction (prefix scan, stable order across replays)
    __shared__ int s_pc[256];
    __shared__ int s_off[257];
    int keep = 0;
#pragma unroll
    for (int i = 0; i < 64; ++i) keep += (u[i] >= thr && u[i] != 0u);
    s_pc[t] = keep;
    __syncthreads();
    if (t == 0) {
        int s = 0;
        for (int i = 0; i < 256; ++i) { s_off[i] = s; s += s_pc[i]; }
        s_off[256] = s;
        counts[row] = s > MAX_NNZ ? MAX_NNZ : s;
    }
    __syncthreads();
    int pos = s_off[t];
#pragma unroll
    for (int i = 0; i < 64; ++i) {
        const unsigned ui = u[i];
        const int col = t + i * 256;
        if (ui >= thr && ui != 0u) {
            if (pos < MAX_NNZ) {
                idxs[(size_t)row * MAX_NNZ + pos] = col;
                vals[(size_t)row * MAX_NNZ + pos] = __uint_as_float(ui);
            }
            ++pos;
        } else {
            __builtin_nontemporal_store(0.f, lrow + col);  // mask out
        }
    }
}

// ---------------------------------------------------------------------------
// 5) Sparse decoder: recon[row] = b_dec + sum_e val_e * W_dec[idx_e, :]
//    W_dec (128 MB) is L2-resident; one workgroup per row.
// ---------------------------------------------------------------------------
__global__ __launch_bounds__(256) void k_decoder(
        const float* __restrict__ Wd,     // [LATENT_DIM][INPUT_DIM]
        const float* __restrict__ b_dec,
        const int*   __restrict__ counts,
        const int*   __restrict__ idxs,
        const float* __restrict__ vals,
        float*       __restrict__ recon,
        float*       __restrict__ extra) {
    const int row = blockIdx.x;
    const int t   = threadIdx.x;
    __shared__ int   s_n;
    __shared__ int   s_i[MAX_NNZ];
    __shared__ float s_v[MAX_NNZ];
    if (t == 0) s_n = counts[row];
    if (t < MAX_NNZ) {
        s_i[t] = idxs[(size_t)row * MAX_NNZ + t];
        s_v[t] = vals[(size_t)row * MAX_NNZ + t];
    }
    __syncthreads();
    const int n = s_n;

    float acc[8];
#pragma unroll
    for (int j = 0; j < 8; ++j) acc[j] = b_dec[t + j * 256];

    for (int e = 0; e < n; ++e) {
        const float v = s_v[e];
        const float* wr = Wd + (size_t)s_i[e] * INPUT_DIM;
        __builtin_prefetch(wr + t, 0, 3);
#pragma unroll
        for (int j = 0; j < 8; ++j) acc[j] += v * wr[t + j * 256];
    }

    float* out = recon + (size_t)row * INPUT_DIM;
#pragma unroll
    for (int j = 0; j < 8; ++j)
        __builtin_nontemporal_store(acc[j], out + t + j * 256);

    if (row == 0 && t == 0) extra[0] = 0.f;   // trailing scalar output (0.0)
}

// ---------------------------------------------------------------------------
extern "C" void kernel_launch(void* const* d_in, const int* in_sizes, int n_in,
                              void* d_out, int out_size, void* d_ws, size_t ws_size,
                              hipStream_t stream) {
    const float* x     = (const float*)d_in[0];
    const float* W_enc = (const float*)d_in[1];
    const float* b_enc = (const float*)d_in[2];
    const float* W_dec = (const float*)d_in[3];
    const float* b_dec = (const float*)d_in[4];

    float* out    = (float*)d_out;
    float* recon  = out;                                      // [BATCH][INPUT_DIM]
    float* latent = out + (size_t)BATCH * INPUT_DIM;          // [BATCH][LATENT_DIM]
    float* extra  = latent + (size_t)BATCH * LATENT_DIM;      // scalar 0.0

    char*   ws  = (char*)d_ws;
    size_t  off = 0;
    __bf16* xb  = (__bf16*)(ws + off); off += (size_t)BATCH * INPUT_DIM * 2;      // 32 MB
    __bf16* wt  = (__bf16*)(ws + off); off += (size_t)LATENT_DIM * INPUT_DIM * 2; // 64 MB
    int*    counts = (int*)(ws + off); off += (size_t)BATCH * 4;
    int*    idxs   = (int*)(ws + off); off += (size_t)BATCH * MAX_NNZ * 4;
    float*  vals   = (float*)(ws + off);

    k_convert_x<<<(BATCH * INPUT_DIM) / (256 * 4), 256, 0, stream>>>(x, xb);
    k_convT_w<<<dim3(LATENT_DIM / 64, INPUT_DIM / 64), 256, 0, stream>>>(W_enc, wt);
    k_encoder_gemm<<<dim3(LATENT_DIM / BN, BATCH / BM), 256, 0, stream>>>(xb, wt, b_enc, latent);
    k_topk<<<BATCH, 256, 0, stream>>>(latent, counts, idxs, vals);
    k_decoder<<<BATCH, 256, 0, stream>>>(W_dec, b_dec, counts, idxs, vals, recon, extra);
}
